// SPGATConv_58394375356694
// MI455X (gfx1250) — matrix operand order repaired and checked by reading, not compiled, not run on hardware
//
#include <hip/hip_runtime.h>
#include <hip/hip_bf16.h>

// Problem constants from the reference (fixed-size dispatch):
//   IN_FEATS=128, H=4, D=16 -> H*D=64, N_total=150000 (divisible by 16), E=1.2M
#define IN_F   128
#define OUT_F  64   // H*D
#define NHEAD  4
#define DHEAD  16

typedef __attribute__((ext_vector_type(16))) _Float16 v16h;
typedef __attribute__((ext_vector_type(8)))  float    v8f;

// ---------- helpers ----------
__device__ __forceinline__ unsigned encf(float x) {
    // order-preserving float -> uint encoding (for atomicMax on floats incl. negatives)
    unsigned u = __float_as_uint(x);
    return (u & 0x80000000u) ? ~u : (u | 0x80000000u);
}
__device__ __forceinline__ float decf(unsigned e) {
    unsigned u = (e & 0x80000000u) ? (e & 0x7FFFFFFFu) : ~e;
    return __uint_as_float(u);
}
__device__ __forceinline__ float lrelu(float x) { return x > 0.0f ? x : 0.2f * x; }

// ---------- 1) pre-swizzle B = W^T (64x128 -> WMMA B fragments, f16) ----------
// B-fragment layout (16-bit, 32x16 KxN, wave32): lane L holds column N = L&15;
// lanes 0-15 carry K = 0..7,16..23 (e<8 -> K=e, else K=e+8), lanes 16-31 carry K = 8..15,24..31.
// Fragment (s = K-step 0..3, b = N-block 0..3) stored contiguously: 32 lanes x 16 halves.
__global__ void prep_b_kernel(const float* __restrict__ W, _Float16* __restrict__ bfrag) {
    int t = blockIdx.x * blockDim.x + threadIdx.x;
    if (t >= 4 * 4 * 32 * 16) return;
    int e = t & 15;
    int L = (t >> 4) & 31;
    int b = (t >> 9) & 3;
    int s = (t >> 11) & 3;
    int n = b * DHEAD + (L & 15);
    int k = s * 32 + ((L >> 4) * 8) + (e & 7) + ((e >> 3) * 16);
    bfrag[t] = (_Float16)W[n * IN_F + k];   // B[k][n] = W[n][k]
}

// ---------- 2) proj = feat @ W^T via v_wmma_f32_16x16x32_f16 ----------
// One wave -> one 16-row tile x 64 cols (4 accumulators). K loop: 4 steps of 32.
// A layout (16-bit 16x32): lane L -> row L&15; half = L>>4; e<8 -> K = half*8+e,
// e>=8 -> K = 16 + half*8 + (e-8): two contiguous 8-float runs per step.
__global__ void __launch_bounds__(256) proj_wmma_kernel(const float* __restrict__ feat,
                                                        const _Float16* __restrict__ bfrag,
                                                        float* __restrict__ proj,
                                                        int n_tiles) {
    int wave = threadIdx.x >> 5;
    int lane = threadIdx.x & 31;
    int tile = blockIdx.x * 8 + wave;
    if (tile >= n_tiles) return;          // uniform per wave: EXEC stays all-ones for WMMA

    int row0 = tile * 16;
    int half = lane >> 4;
    int mrow = row0 + (lane & 15);
    const float* abase = feat + (size_t)mrow * IN_F + half * 8;

    v8f c0 = {}, c1 = {}, c2 = {}, c3 = {};
    const v16h* bp = (const v16h*)bfrag;

#pragma unroll
    for (int s = 0; s < 4; ++s) {
        const float* ab = abase + s * 32;
        float4 f0 = *(const float4*)(ab);
        float4 f1 = *(const float4*)(ab + 4);
        float4 f2 = *(const float4*)(ab + 16);
        float4 f3 = *(const float4*)(ab + 20);
        v16h a;
        a[0]=(_Float16)f0.x;  a[1]=(_Float16)f0.y;  a[2]=(_Float16)f0.z;  a[3]=(_Float16)f0.w;
        a[4]=(_Float16)f1.x;  a[5]=(_Float16)f1.y;  a[6]=(_Float16)f1.z;  a[7]=(_Float16)f1.w;
        a[8]=(_Float16)f2.x;  a[9]=(_Float16)f2.y;  a[10]=(_Float16)f2.z; a[11]=(_Float16)f2.w;
        a[12]=(_Float16)f3.x; a[13]=(_Float16)f3.y; a[14]=(_Float16)f3.z; a[15]=(_Float16)f3.w;

        v16h b0 = bp[(s * 4 + 0) * 32 + lane];
        v16h b1 = bp[(s * 4 + 1) * 32 + lane];
        v16h b2 = bp[(s * 4 + 2) * 32 + lane];
        v16h b3 = bp[(s * 4 + 3) * 32 + lane];

        c0 = __builtin_amdgcn_wmma_f32_16x16x32_f16(false, a, false, b0, (short)0, c0, false, false);
        c1 = __builtin_amdgcn_wmma_f32_16x16x32_f16(false, a, false, b1, (short)0, c1, false, false);
        c2 = __builtin_amdgcn_wmma_f32_16x16x32_f16(false, a, false, b2, (short)0, c2, false, false);
        c3 = __builtin_amdgcn_wmma_f32_16x16x32_f16(false, a, false, b3, (short)0, c3, false, false);
    }

    // C/D layout: lane L col N = L&15; VGPR v -> row M = v + half*8
    int ncol = lane & 15;
#pragma unroll
    for (int v = 0; v < 8; ++v) {
        float* pr = proj + (size_t)(row0 + half * 8 + v) * OUT_F + ncol;
        pr[0]  = c0[v];
        pr[16] = c1[v];
        pr[32] = c2[v];
        pr[48] = c3[v];
    }
}

// ---------- 3) per-node scores: er (rows < num_dst, attn_r) / el (src rows, attn_l) ----------
__global__ void score_kernel(const float* __restrict__ proj,
                             const float* __restrict__ attn_l,
                             const float* __restrict__ attn_r,
                             float* __restrict__ score, int n_total, int num_dst) {
    int t = blockIdx.x * blockDim.x + threadIdx.x;
    if (t >= n_total * NHEAD) return;
    int n = t >> 2, h = t & 3;
    const float* attn = (n < num_dst) ? attn_r : attn_l;
    const float* p = proj + (size_t)n * OUT_F + h * DHEAD;
    float s = 0.0f;
#pragma unroll
    for (int d = 0; d < DHEAD; ++d) s += p[d] * attn[h * DHEAD + d];
    score[t] = s;
}

// ---------- 4) init: out = bias, m = sentinel(0), denom = 0 ----------
__global__ void init_kernel(float* __restrict__ out, const float* __restrict__ bias,
                            unsigned* __restrict__ m_enc, float* __restrict__ denom,
                            int num_dst) {
    int t = blockIdx.x * blockDim.x + threadIdx.x;
    if (t < num_dst * OUT_F) out[t] = bias[t & (OUT_F - 1)];
    if (t < num_dst * NHEAD) { m_enc[t] = 0u; denom[t] = 0.0f; }
}

// ---------- 5) softmax pass 1: segment max (uint-encoded atomic max) ----------
__global__ void edge_max_kernel(const int* __restrict__ esrc, const int* __restrict__ edst,
                                const float* __restrict__ score,
                                unsigned* __restrict__ m_enc, int E, int num_dst) {
    int t = blockIdx.x * blockDim.x + threadIdx.x;
    if (t >= E * NHEAD) return;
    int e = t >> 2, h = t & 3;
    int s = esrc[e], d = edst[e];
    float x = lrelu(score[(num_dst + s) * NHEAD + h] + score[d * NHEAD + h]);
    atomicMax(&m_enc[d * NHEAD + h], encf(x));
}

// ---------- 6) softmax pass 2: segment sum of exp(e - m) ----------
__global__ void edge_sum_kernel(const int* __restrict__ esrc, const int* __restrict__ edst,
                                const float* __restrict__ score,
                                const unsigned* __restrict__ m_enc,
                                float* __restrict__ denom, int E, int num_dst) {
    int t = blockIdx.x * blockDim.x + threadIdx.x;
    if (t >= E * NHEAD) return;
    int e = t >> 2, h = t & 3;
    int s = esrc[e], d = edst[e];
    float x = lrelu(score[(num_dst + s) * NHEAD + h] + score[d * NHEAD + h]);
    float m = decf(m_enc[d * NHEAD + h]);
    atomicAdd(&denom[d * NHEAD + h], expf(x - m));
}

// ---------- 7) aggregate: one wave per edge, lane covers 2 of 64 feature cols ----------
__global__ void edge_agg_kernel(const int* __restrict__ esrc, const int* __restrict__ edst,
                                const float* __restrict__ score,
                                const unsigned* __restrict__ m_enc,
                                const float* __restrict__ denom,
                                const float* __restrict__ proj,
                                float* __restrict__ out, int E, int num_dst) {
    int wave = (int)((blockIdx.x * (unsigned)blockDim.x + threadIdx.x) >> 5);
    int lane = threadIdx.x & 31;
    if (wave >= E) return;
    int s = esrc[wave], d = edst[wave];
    int j = lane * 2;          // feature columns j, j+1
    int h = j >> 4;            // head of these columns
    float x = lrelu(score[(num_dst + s) * NHEAD + h] + score[d * NHEAD + h]);
    float m = decf(m_enc[d * NHEAD + h]);
    float a = expf(x - m) / denom[d * NHEAD + h];
    float2 v = *(const float2*)(proj + (size_t)(num_dst + s) * OUT_F + j);
    float* o = out + (size_t)d * OUT_F + j;
    atomicAdd(o,     v.x * a);
    atomicAdd(o + 1, v.y * a);
}

extern "C" void kernel_launch(void* const* d_in, const int* in_sizes, int n_in,
                              void* d_out, int out_size, void* d_ws, size_t ws_size,
                              hipStream_t stream) {
    const float* feat   = (const float*)d_in[0];
    const float* W      = (const float*)d_in[1];
    const float* attn_l = (const float*)d_in[2];
    const float* attn_r = (const float*)d_in[3];
    const float* bias   = (const float*)d_in[4];
    const int*   esrc   = (const int*)d_in[5];
    const int*   edst   = (const int*)d_in[6];

    const int E       = in_sizes[5];
    const int n_total = in_sizes[0] / IN_F;   // 150000
    const int num_dst = out_size / OUT_F;     // 50000 (derived from output shape)
    float* out = (float*)d_out;

    // workspace carve-up
    char* ws = (char*)d_ws;
    float*    proj  = (float*)ws;
    float*    score = proj + (size_t)n_total * OUT_F;
    unsigned* m_enc = (unsigned*)(score + (size_t)n_total * NHEAD);
    float*    denom = (float*)(m_enc + (size_t)num_dst * NHEAD);
    uintptr_t bp    = (uintptr_t)(denom + (size_t)num_dst * NHEAD);
    bp = (bp + 31u) & ~(uintptr_t)31u;        // 32B align for v16h loads
    _Float16* bfrag = (_Float16*)bp;

    const int T = 256;
    prep_b_kernel<<<(4 * 4 * 32 * 16 + T - 1) / T, T, 0, stream>>>(W, bfrag);

    int n_tiles = n_total / 16;               // 150000 is a multiple of 16
    proj_wmma_kernel<<<(n_tiles + 7) / 8, T, 0, stream>>>(feat, bfrag, proj, n_tiles);

    score_kernel<<<(n_total * NHEAD + T - 1) / T, T, 0, stream>>>(proj, attn_l, attn_r,
                                                                  score, n_total, num_dst);
    init_kernel<<<(num_dst * OUT_F + T - 1) / T, T, 0, stream>>>(out, bias, m_enc, denom, num_dst);
    edge_max_kernel<<<(E * NHEAD + T - 1) / T, T, 0, stream>>>(esrc, edst, score, m_enc, E, num_dst);
    edge_sum_kernel<<<(E * NHEAD + T - 1) / T, T, 0, stream>>>(esrc, edst, score, m_enc, denom, E, num_dst);
    edge_agg_kernel<<<(E + 7) / 8, T, 0, stream>>>(esrc, edst, score, m_enc, denom, proj, out, E, num_dst);
}
